// MSDeformAttn_with_GlobalRegisters_9680856285303
// MI455X (gfx1250) — compile-verified, matrix operand order checked by reading, not run on hardware
//
#include <hip/hip_runtime.h>
#include <hip/hip_bf16.h>

#define BATCH 4
#define LQ    21792
#define DM    256
#define NH    8
#define HD    32
#define NL    4
#define NP    4
#define NR    8
#define MROWS (BATCH * LQ)           // 87168, divisible by 32
#define NELEM ((size_t)MROWS * DM)   // 22,315,008

typedef __attribute__((ext_vector_type(16))) _Float16 v16h;
typedef __attribute__((ext_vector_type(8)))  _Float16 v8h;
typedef __attribute__((ext_vector_type(8)))  float    v8f;

union H16 { v16h v; struct { v8h lo; v8h hi; } s; };

// ---------------------------------------------------------------------------
// f32 -> f16 elementwise conversion, 8 elements/thread, b128 stores.
// n8 = n/8 (all our sizes are multiples of 8).
__global__ __launch_bounds__(256) void cvt_f16(
    const float* __restrict__ in, _Float16* __restrict__ out, int n8)
{
    int i = blockIdx.x * 256 + threadIdx.x;
    if (i >= n8) return;
    const float4* p = (const float4*)in + (size_t)i * 2;
    float4 a = p[0], b = p[1];
    v8h o;
    o[0] = (_Float16)a.x; o[1] = (_Float16)a.y; o[2] = (_Float16)a.z; o[3] = (_Float16)a.w;
    o[4] = (_Float16)b.x; o[5] = (_Float16)b.y; o[6] = (_Float16)b.z; o[7] = (_Float16)b.w;
    *((v8h*)out + i) = o;
}

// reg_w is [256(K), 256(N)] f32 -> regWT f16 [256(N), 256(K)], coalesced writes.
__global__ __launch_bounds__(256) void transpose_regw_cvt(
    const float* __restrict__ in, _Float16* __restrict__ out)
{
    int idx = blockIdx.x * 256 + threadIdx.x;   // idx = n*256 + c
    int n = idx >> 8, c = idx & 255;
    out[idx] = (_Float16)in[c * 256 + n];
}

// ---------------------------------------------------------------------------
// C[M, Nn] = A[M, 256] @ Bw[Nn, 256]^T + bias[Nn]   (A, Bw in f16; C f32)
// Block = 256 threads = 8 waves. Wave w: M sub-tile (w&1), 64-col quad (w>>1).
// Each wave: one A fragment reused across 4 WMMAs per k-step (4 acc chains).
// A-fragment (16x32 f16): lane m in [0,15] -> row m, K {0..7,16..23};
// lanes 16..31 -> rows 0..15, K {8..15,24..31}; both chunks contiguous b128.
__global__ __launch_bounds__(256) void gemm_xwt_wmma(
    const _Float16* __restrict__ A, const _Float16* __restrict__ Bw,
    const float* __restrict__ bias, float* __restrict__ C, int Nn)
{
    const int K   = DM;
    int lane = threadIdx.x & 31;
    int w    = threadIdx.x >> 5;
    int quad = w >> 1;                       // 64-column quad
    if (quad * 64 >= Nn) return;             // wave-uniform (Nn=128 case)
    int mt   = blockIdx.x * 2 + (w & 1);     // 16-row tile

    int l15   = lane & 15;
    int hi    = lane >> 4;
    int khalf = hi << 3;                     // 0 or 8
    const _Float16* arow = A + (size_t)(mt * 16 + l15) * K + khalf;
    const _Float16* brow0 = Bw + (size_t)(quad * 64 + l15) * K + khalf;

    v8f acc0 = {}, acc1 = {}, acc2 = {}, acc3 = {};
#pragma unroll
    for (int k0 = 0; k0 < K; k0 += 32) {
        H16 af, b0, b1, b2, b3;
        af.s.lo = *(const v8h*)(arow + k0);
        af.s.hi = *(const v8h*)(arow + k0 + 16);
        b0.s.lo = *(const v8h*)(brow0 + k0);
        b0.s.hi = *(const v8h*)(brow0 + k0 + 16);
        b1.s.lo = *(const v8h*)(brow0 + 16 * K + k0);
        b1.s.hi = *(const v8h*)(brow0 + 16 * K + k0 + 16);
        b2.s.lo = *(const v8h*)(brow0 + 32 * K + k0);
        b2.s.hi = *(const v8h*)(brow0 + 32 * K + k0 + 16);
        b3.s.lo = *(const v8h*)(brow0 + 48 * K + k0);
        b3.s.hi = *(const v8h*)(brow0 + 48 * K + k0 + 16);
        acc0 = __builtin_amdgcn_wmma_f32_16x16x32_f16(false, af.v, false, b0.v, (short)0, acc0, false, false);
        acc1 = __builtin_amdgcn_wmma_f32_16x16x32_f16(false, af.v, false, b1.v, (short)0, acc1, false, false);
        acc2 = __builtin_amdgcn_wmma_f32_16x16x32_f16(false, af.v, false, b2.v, (short)0, acc2, false, false);
        acc3 = __builtin_amdgcn_wmma_f32_16x16x32_f16(false, af.v, false, b3.v, (short)0, acc3, false, false);
    }

    // D layout: VGPR r -> row mt*16 + 8*hi + r; col = tile base + (lane&15).
    float* crow = C + (size_t)(mt * 16 + (hi << 3)) * Nn + quad * 64 + l15;
    v8f accs[4] = {acc0, acc1, acc2, acc3};
#pragma unroll
    for (int t = 0; t < 4; ++t) {
        float bv = bias[quad * 64 + t * 16 + l15];
#pragma unroll
        for (int r = 0; r < 8; ++r)
            crow[(size_t)r * Nn + t * 16] = accs[t][r] + bv;
    }
}

// ---------------------------------------------------------------------------
// In-place: so logits [bq][h,l,p,c] -> sampling_locations = ref + so / norm
__global__ __launch_bounds__(256) void loc_transform(
    float* __restrict__ so, const float* __restrict__ ref, const int* __restrict__ shapes)
{
    int idx = blockIdx.x * 256 + threadIdx.x;   // [bq][256]
    int n  = idx & 255;
    int bq = idx >> 8;
    int c  = n & 1;
    int l  = (n >> 3) & 3;
    float norm = (float)(c == 0 ? shapes[l * 2 + 1] : shapes[l * 2]); // W or H
    so[idx] = ref[(size_t)bq * 8 + l * 2 + c] + so[idx] / norm;
}

// In-place softmax over rows of 16 (aw over L*P)
__global__ __launch_bounds__(256) void softmax16(float* __restrict__ a)
{
    int row = blockIdx.x * 256 + threadIdx.x;   // B*LQ*NH rows
    float* p = a + (size_t)row * 16;
    float m = p[0];
#pragma unroll
    for (int i = 1; i < 16; ++i) m = fmaxf(m, p[i]);
    float e[16], s = 0.f;
#pragma unroll
    for (int i = 0; i < 16; ++i) { e[i] = __expf(p[i] - m); s += e[i]; }
    float inv = 1.f / s;
#pragma unroll
    for (int i = 0; i < 16; ++i) p[i] = e[i] * inv;
}

// ---------------------------------------------------------------------------
// Fused bilinear deformable sampling + register branch.
// One wave per (b,q,h); lanes = 32 head channels -> 128B coalesced gathers.
// mid = 0.7*local + 0.3*reg_out.
__global__ __launch_bounds__(256) void sample_reg(
    const float* __restrict__ val,   // [b, vq, h, d] = [bq][256]
    const float* __restrict__ loc,   // sampling_locations [bq][h,l,p,c]
    const float* __restrict__ aw,    // softmaxed [bq*NH][16]
    const float* __restrict__ rw,    // rw logits [bq][h*32+i]
    const int*   __restrict__ shapes,
    float* __restrict__ mid)         // [bq][256]
{
    const int hwstart[4] = {8, 16400, 20504, 21536};
    const int regbase[4] = {0, 16392, 20496, 21528};

    int lane = threadIdx.x & 31;                       // channel d
    int g    = blockIdx.x * 8 + (threadIdx.x >> 5);    // g = bq*8 + h
    int b    = g / (LQ * NH);
    int h    = g & 7;
    const float* vb = val + (size_t)b * LQ * DM + h * HD + lane;

    // ---- local (deformable sampling) branch ----
    float acc = 0.f;
#pragma unroll
    for (int l = 0; l < NL; ++l) {
        int Hl = shapes[l * 2], Wl = shapes[l * 2 + 1];
        int base = hwstart[l];
#pragma unroll
        for (int p = 0; p < NP; ++p) {
            float lx = loc[(size_t)g * 32 + l * 8 + p * 2 + 0];
            float ly = loc[(size_t)g * 32 + l * 8 + p * 2 + 1];
            float w  = aw[(size_t)g * 16 + l * 4 + p];
            float x  = lx * (float)Wl - 0.5f;
            float y  = ly * (float)Hl - 0.5f;
            float fx = floorf(x), fy = floorf(y);
            float dx = x - fx,    dy = y - fy;
            int   x0 = (int)fx,   y0 = (int)fy;
#pragma unroll
            for (int cy = 0; cy < 2; ++cy) {
                int   yi = y0 + cy;
                float wy = cy ? dy : 1.f - dy;
                bool  vy = (yi >= 0) && (yi < Hl);
                int   yc = min(max(yi, 0), Hl - 1);
#pragma unroll
                for (int cx = 0; cx < 2; ++cx) {
                    int   xi = x0 + cx;
                    float wx = cx ? dx : 1.f - dx;
                    bool  vx = (xi >= 0) && (xi < Wl);
                    int   xc = min(max(xi, 0), Wl - 1);
                    float wt = (vx && vy) ? wx * wy * w : 0.f;
                    acc += wt * vb[(size_t)(base + yc * Wl + xc) * DM];
                }
            }
        }
    }

    // ---- register (global) branch: softmax over 32 then matvec ----
    const float* rwp = rw + (size_t)g * 32;
    float mx = rwp[0];
#pragma unroll
    for (int i = 1; i < 32; ++i) mx = fmaxf(mx, rwp[i]);
    float e[32], s = 0.f;
#pragma unroll
    for (int i = 0; i < 32; ++i) { e[i] = __expf(rwp[i] - mx); s += e[i]; }
    float inv = 1.f / s;
    float racc = 0.f;
#pragma unroll
    for (int i = 0; i < 32; ++i) {
        int l = i >> 3, r = i & 7;
        racc += (e[i] * inv) * vb[(size_t)(regbase[l] + r) * DM];
    }

    mid[(size_t)g * 32 + lane] = 0.7f * acc + 0.3f * racc;
}

// ---------------------------------------------------------------------------
extern "C" void kernel_launch(void* const* d_in, const int* in_sizes, int n_in,
                              void* d_out, int out_size, void* d_ws, size_t ws_size,
                              hipStream_t stream)
{
    (void)in_sizes; (void)n_in; (void)out_size; (void)ws_size;

    const float* query  = (const float*)d_in[0];
    const float* refp   = (const float*)d_in[1];
    const float* inflat = (const float*)d_in[2];
    const int*   shapes = (const int*)d_in[3];
    /* d_in[4] level_start_index: recomputed statically */
    const float* vp_w = (const float*)d_in[5];
    const float* vp_b = (const float*)d_in[6];
    const float* so_w = (const float*)d_in[7];
    const float* so_b = (const float*)d_in[8];
    const float* aw_w = (const float*)d_in[9];
    const float* aw_b = (const float*)d_in[10];
    const float* regw = (const float*)d_in[11];
    const float* regb = (const float*)d_in[12];
    const float* op_w = (const float*)d_in[13];
    const float* op_b = (const float*)d_in[14];

    float* out0    = (float*)d_out;                       // output [MROWS,256]
    float* out_loc = out0    + NELEM;                     // sampling_locations
    float* out_aw  = out_loc + NELEM;                     // attention weights

    float* ws   = (float*)d_ws;
    float* valF = ws;                 // [MROWS,256] f32 (sampling precision)
    float* midF = valF + NELEM;       // [MROWS,256] f32
    float* rwF  = midF + NELEM;       // [MROWS,256] f32
    _Float16* qH  = (_Float16*)(rwF + NELEM);  // query f16
    _Float16* hB  = qH + NELEM;                // shared: inflat f16, then mid f16
    _Float16* vpH = hB + NELEM;
    _Float16* soH = vpH + 256 * 256;
    _Float16* awH = soH + 256 * 256;
    _Float16* rgH = awH + 128 * 256;           // transposed reg_w, f16
    _Float16* opH = rgH + 256 * 256;

    dim3 blk(256);
    const int actBlks = (int)(NELEM / 8 / 256);   // 10896
    const int w64Blks = 65536 / 8 / 256;          // 32
    const int w32Blks = 32768 / 8 / 256;          // 16

    // f32 -> f16 staging (once per element)
    cvt_f16<<<actBlks, blk, 0, stream>>>(query,  qH,  (int)(NELEM / 8));
    cvt_f16<<<actBlks, blk, 0, stream>>>(inflat, hB,  (int)(NELEM / 8));
    cvt_f16<<<w64Blks, blk, 0, stream>>>(vp_w, vpH, 65536 / 8);
    cvt_f16<<<w64Blks, blk, 0, stream>>>(so_w, soH, 65536 / 8);
    cvt_f16<<<w32Blks, blk, 0, stream>>>(aw_w, awH, 32768 / 8);
    cvt_f16<<<w64Blks, blk, 0, stream>>>(op_w, opH, 65536 / 8);
    transpose_regw_cvt<<<256, blk, 0, stream>>>(regw, rgH);

    // WMMA GEMMs: block covers 32 rows x 256 cols
    dim3 gg(MROWS / 32);   // 2724
    gemm_xwt_wmma<<<gg, blk, 0, stream>>>(hB, vpH, vp_b, valF,    256);
    gemm_xwt_wmma<<<gg, blk, 0, stream>>>(qH, soH, so_b, out_loc, 256);
    gemm_xwt_wmma<<<gg, blk, 0, stream>>>(qH, awH, aw_b, out_aw,  128);
    gemm_xwt_wmma<<<gg, blk, 0, stream>>>(qH, rgH, regb, rwF,     256);

    loc_transform<<<MROWS, blk, 0, stream>>>(out_loc, refp, shapes);
    softmax16<<<(MROWS * NH) / 256, blk, 0, stream>>>(out_aw);

    sample_reg<<<MROWS, blk, 0, stream>>>(valF, out_loc, out_aw, rwF, shapes, midF);

    // mid f32 -> f16 (reuses inflat's f16 buffer), then output projection
    cvt_f16<<<actBlks, blk, 0, stream>>>(midF, hB, (int)(NELEM / 8));
    gemm_xwt_wmma<<<gg, blk, 0, stream>>>(hB, opH, op_b, out0, 256);
}
